// VariationalGCNEncoder_50577534877935
// MI455X (gfx1250) — matrix operand order repaired and checked by reading, not compiled, not run on hardware
//
#include <hip/hip_runtime.h>

#define NODES 100000
#define EDGES 1600000
#define EN_TOT (EDGES + NODES)
#define CIN 128
#define NEG_SLOPE 0.2f

typedef __attribute__((ext_vector_type(16))) _Float16 v16h;
typedef __attribute__((ext_vector_type(8)))  _Float16 v8h;
typedef __attribute__((ext_vector_type(8)))  float    v8f;

// ---- order-preserving float<->uint encoding for atomic segment-max ----
__device__ __forceinline__ unsigned fenc(float f) {
  unsigned b = __float_as_uint(f);
  return (b & 0x80000000u) ? ~b : (b | 0x80000000u);
}
__device__ __forceinline__ float fdec(unsigned e) {
  unsigned b = (e & 0x80000000u) ? (e ^ 0x80000000u) : ~e;
  return __uint_as_float(b);
}

// ---------------- elementwise helpers ----------------
__global__ void k_fill_f32(float* __restrict__ p, float v, long n) {
  long i = (long)blockIdx.x * blockDim.x + threadIdx.x;
  if (i < n) p[i] = v;
}

__global__ void k_cvt_f16(const float* __restrict__ in, _Float16* __restrict__ out, long n) {
  long i = (long)blockIdx.x * blockDim.x + threadIdx.x;
  if (i < n) out[i] = (_Float16)in[i];
}

// Pre-swizzle W [CIN x cout] (f32, row-major) into B-fragment order (f16):
// storage index = (((j*4 + c)*32 + lane)*16 + e), element (K,N) with
// K = 32c + 16*(lane>=16) + e, N = 16j + (lane&15).
__global__ void k_swizzle_w(const float* __restrict__ W, _Float16* __restrict__ Wf, int cout) {
  int t = blockIdx.x * blockDim.x + threadIdx.x;
  if (t >= CIN * cout) return;
  int e = t & 15;
  int l = (t >> 4) & 31;
  int rem = t >> 9;
  int c = rem & 3;      // CIN/32 == 4
  int j = rem >> 2;
  int K  = 32 * c + ((l >= 16) ? 16 : 0) + e;
  int nn = 16 * j + (l & 15);
  Wf[t] = (_Float16)W[K * cout + nn];
}

// dot of two small vectors (n <= 128) -> out[0]   (edge-attn scalar c_e = We . a_e)
__global__ void k_dot(const float* __restrict__ u, const float* __restrict__ v,
                      float* __restrict__ out, int n) {
  __shared__ float sm[128];
  int t = threadIdx.x;
  sm[t] = (t < n) ? u[t] * v[t] : 0.f;
  __syncthreads();
  for (int s = 64; s > 0; s >>= 1) { if (t < s) sm[t] += sm[t + s]; __syncthreads(); }
  if (t == 0) out[0] = sm[0];
}

// in-degree count and edge-weight sum per destination (for fill_value='mean')
__global__ void k_deg(const int* __restrict__ dst, const float* __restrict__ ew,
                      float* __restrict__ cnt, float* __restrict__ sumw) {
  int i = blockIdx.x * blockDim.x + threadIdx.x;
  if (i >= EDGES) return;
  int d = dst[i];
  atomicAdd(&cnt[d], 1.f);
  atomicAdd(&sumw[d], ew[i]);
}

__global__ void k_loopattr(const float* __restrict__ cnt, const float* __restrict__ sumw,
                           float* __restrict__ lattr) {
  int i = blockIdx.x * blockDim.x + threadIdx.x;
  if (i >= NODES) return;
  float c = cnt[i];
  lattr[i] = (c > 0.f) ? (sumw[i] / c) : 0.f;
}

// ---------------- WMMA GEMM: XL[N x COUT] = H16[N x 128] * Wf ----------------
// one wave per 16-row strip; K = 128 in four 16x16x32 WMMA steps per col tile.
template <int COUT>
__global__ void k_gemm_wmma(const _Float16* __restrict__ H, const _Float16* __restrict__ Wf,
                            float* __restrict__ XL) {
  const int NSTRIPS = NODES / 16;  // 6250, exact
  int wave = blockIdx.x * (blockDim.x >> 5) + (threadIdx.x >> 5);
  if (wave >= NSTRIPS) return;     // wave-uniform branch; surviving waves have full EXEC
  int lane = threadIdx.x & 31;
  int row0 = wave * 16;
  int klo  = (lane >= 16) ? 16 : 0;
  const _Float16* hrow = H + (size_t)(row0 + (lane & 15)) * CIN;

  // A fragments (contiguous k order, matching Wf): element e -> K=base+16*(lane>=16)+e
  v16h a[4];
#pragma unroll
  for (int c = 0; c < 4; c++) {
    v8h lo = *(const v8h*)(hrow + 32 * c + klo);
    v8h hi = *(const v8h*)(hrow + 32 * c + klo + 8);
    a[c] = __builtin_shufflevector(lo, hi, 0, 1, 2, 3, 4, 5, 6, 7, 8, 9, 10, 11, 12, 13, 14, 15);
  }

  int col   = (lane & 15);
  int rbase = row0 + ((lane >= 16) ? 8 : 0);
#pragma unroll
  for (int j = 0; j < COUT / 16; j++) {
    v8f acc = {0.f, 0.f, 0.f, 0.f, 0.f, 0.f, 0.f, 0.f};
#pragma unroll
    for (int c = 0; c < 4; c++) {
      const _Float16* bp = Wf + ((size_t)((j * 4 + c) * 32 + lane)) * 16;
      v8h lo = *(const v8h*)(bp);
      v8h hi = *(const v8h*)(bp + 8);
      v16h b = __builtin_shufflevector(lo, hi, 0, 1, 2, 3, 4, 5, 6, 7, 8, 9, 10, 11, 12, 13, 14, 15);
      acc = __builtin_amdgcn_wmma_f32_16x16x32_f16(false, a[c], false, b, (short)0, acc,
                                                   false, false);
    }
#pragma unroll
    for (int r = 0; r < 8; r++)
      XL[(size_t)(rbase + r) * COUT + j * 16 + col] = acc[r];
  }
}

// per-node attention logits: asrc[n] = xl[n,:].a_s ; adst[n] = xl[n,:].a_d
template <int COUT>
__global__ void k_logits(const float* __restrict__ XL, const float* __restrict__ avs,
                         const float* __restrict__ avd, float* __restrict__ asrc,
                         float* __restrict__ adst) {
  int node = blockIdx.x * (blockDim.x >> 5) + (threadIdx.x >> 5);
  if (node >= NODES) return;
  int lane = threadIdx.x & 31;
  const int CPL = COUT / 32;
  const float* row = XL + (size_t)node * COUT + lane * CPL;
  float s = 0.f, d = 0.f;
#pragma unroll
  for (int k = 0; k < CPL; k++) {
    float xv = row[k];
    s += xv * avs[lane * CPL + k];
    d += xv * avd[lane * CPL + k];
  }
#pragma unroll
  for (int o = 16; o > 0; o >>= 1) { s += __shfl_xor(s, o, 32); d += __shfl_xor(d, o, 32); }
  if (lane == 0) { asrc[node] = s; adst[node] = d; }
}

// pass A: alpha = leaky_relu(a_src[s]+a_dst[d]+edge_term); segment-max via encoded atomicMax
__global__ void k_passA(const int* __restrict__ src, const int* __restrict__ dst,
                        const float* __restrict__ ew, const float* __restrict__ asrc,
                        const float* __restrict__ adst, const float* __restrict__ lattr,
                        const float* __restrict__ ce, int has_edge,
                        float* __restrict__ alpha, unsigned* __restrict__ menc) {
  int i = blockIdx.x * blockDim.x + threadIdx.x;
  if (i >= EN_TOT) return;
  int s, d; float e = 0.f;
  if (i < EDGES) { s = src[i]; d = dst[i]; if (has_edge) e = ew[i]; }
  else           { s = d = i - EDGES;      if (has_edge) e = lattr[i - EDGES]; }
  float al = asrc[s] + adst[d] + (has_edge ? e * ce[0] : 0.f);
  al = (al > 0.f) ? al : NEG_SLOPE * al;
  alpha[i] = al;
  atomicMax(&menc[d], fenc(al));
}

// pass B: ex = exp(alpha - m[d]); den[d] += ex; store ex back in alpha
__global__ void k_passB(const int* __restrict__ dst, float* __restrict__ alpha,
                        const unsigned* __restrict__ menc, float* __restrict__ den) {
  int i = blockIdx.x * blockDim.x + threadIdx.x;
  if (i >= EN_TOT) return;
  int d = (i < EDGES) ? dst[i] : (i - EDGES);
  float ex = __expf(alpha[i] - fdec(menc[d]));
  alpha[i] = ex;
  atomicAdd(&den[d], ex);
}

// pass C: acc[d,:] += (ex/den[d]) * xl[s,:]   (one wave per edge, COUT/32 cols per lane)
template <int COUT>
__global__ void k_passC(const int* __restrict__ src, const int* __restrict__ dst,
                        const float* __restrict__ alpha, const float* __restrict__ den,
                        const float* __restrict__ XL, float* __restrict__ acc) {
  int i = blockIdx.x * (blockDim.x >> 5) + (threadIdx.x >> 5);
  if (i >= EN_TOT) return;
  int lane = threadIdx.x & 31;
  int s, d;
  if (i < EDGES) { s = src[i]; d = dst[i]; } else { s = d = i - EDGES; }
  float coef = alpha[i] / den[d];
  const int CPL = COUT / 32;
  const float* xs = XL + (size_t)s * COUT + lane * CPL;
  float* od = acc + (size_t)d * COUT + lane * CPL;
#pragma unroll
  for (int k = 0; k < CPL; k++) atomicAdd(&od[k], coef * xs[k]);
}

// finalize: add bias, optional relu, write f32 output and/or f16 staging for next layer
template <int COUT>
__global__ void k_finalize(const float* __restrict__ acc, const float* __restrict__ bias,
                           float* __restrict__ fout, _Float16* __restrict__ hout, int do_relu) {
  long i = (long)blockIdx.x * blockDim.x + threadIdx.x;
  if (i >= (long)NODES * COUT) return;
  int col = (int)(i & (COUT - 1));
  float v = acc[i] + bias[col];
  if (do_relu) v = fmaxf(v, 0.f);
  if (fout) fout[i] = v;
  if (hout) hout[i] = (_Float16)v;
}

// ---------------- host-side per-layer driver ----------------
template <int COUT>
static void run_layer(const float* W, const float* avs, const float* avd, const float* bias,
                      const float* We, const float* ae, int has_edge, int do_relu,
                      const int* src, const int* dst, const float* ew,
                      _Float16* h16, _Float16* wfrag, float* ce,
                      float* xl, float* agg, float* asrc, float* adst,
                      unsigned* menc, float* den, const float* lattr, float* alpha,
                      float* fout, _Float16* hout, hipStream_t stream) {
  const int TB = 256;
  auto blk = [](long n) { return (unsigned)((n + 255) / 256); };
  k_swizzle_w<<<blk((long)CIN * COUT), TB, 0, stream>>>(W, wfrag, COUT);
  if (has_edge) k_dot<<<1, 128, 0, stream>>>(We, ae, ce, COUT);
  k_gemm_wmma<COUT><<<(NODES / 16 + 7) / 8, TB, 0, stream>>>(h16, wfrag, xl);
  k_logits<COUT><<<(NODES + 7) / 8, TB, 0, stream>>>(xl, avs, avd, asrc, adst);
  k_fill_f32<<<blk(NODES), TB, 0, stream>>>(den, 0.f, NODES);
  k_fill_f32<<<blk(NODES), TB, 0, stream>>>((float*)menc, 0.f, NODES);  // enc-min is bit 0
  k_passA<<<blk(EN_TOT), TB, 0, stream>>>(src, dst, ew, asrc, adst, lattr, ce, has_edge,
                                          alpha, menc);
  k_passB<<<blk(EN_TOT), TB, 0, stream>>>(dst, alpha, menc, den);
  k_fill_f32<<<blk((long)NODES * COUT), TB, 0, stream>>>(agg, 0.f, (long)NODES * COUT);
  k_passC<COUT><<<(EN_TOT + 7) / 8, TB, 0, stream>>>(src, dst, alpha, den, xl, agg);
  k_finalize<COUT><<<blk((long)NODES * COUT), TB, 0, stream>>>(agg, bias, fout, hout, do_relu);
}

extern "C" void kernel_launch(void* const* d_in, const int* in_sizes, int n_in,
                              void* d_out, int out_size, void* d_ws, size_t ws_size,
                              hipStream_t stream) {
  (void)in_sizes; (void)n_in; (void)out_size; (void)ws_size;
  const float* x  = (const float*)d_in[0];
  const int*   ei = (const int*)d_in[1];
  const float* ew = (const float*)d_in[2];
  const float *W0 = (const float*)d_in[3],  *as0 = (const float*)d_in[4],
              *ad0 = (const float*)d_in[5], *ae0 = (const float*)d_in[6],
              *We0 = (const float*)d_in[7], *b0  = (const float*)d_in[8];
  const float *W1 = (const float*)d_in[9],  *as1 = (const float*)d_in[10],
              *ad1 = (const float*)d_in[11], *ae1 = (const float*)d_in[12],
              *We1 = (const float*)d_in[13], *b1  = (const float*)d_in[14];
  const float *Wm = (const float*)d_in[15], *asM = (const float*)d_in[16],
              *adM = (const float*)d_in[17], *aeM = (const float*)d_in[18],
              *WeM = (const float*)d_in[19], *bM  = (const float*)d_in[20];
  const float *Wl = (const float*)d_in[21], *asL = (const float*)d_in[22],
              *adL = (const float*)d_in[23], *bL  = (const float*)d_in[24];
  const int* src = ei;
  const int* dst = ei + EDGES;
  float* outF = (float*)d_out;

  // workspace carve-up (~136 MB)
  char* base = (char*)d_ws; size_t off = 0;
  auto alloc = [&](size_t bytes) -> void* {
    void* p = base + off; off = (off + bytes + 255) & ~(size_t)255; return p;
  };
  _Float16* h16   = (_Float16*)alloc((size_t)NODES * CIN * sizeof(_Float16));
  float*    xl    = (float*)alloc((size_t)NODES * CIN * sizeof(float));
  float*    agg   = (float*)alloc((size_t)NODES * CIN * sizeof(float));
  float*    asrc  = (float*)alloc((size_t)NODES * sizeof(float));
  float*    adstv = (float*)alloc((size_t)NODES * sizeof(float));
  unsigned* menc  = (unsigned*)alloc((size_t)NODES * sizeof(unsigned));
  float*    den   = (float*)alloc((size_t)NODES * sizeof(float));
  float*    cnt   = (float*)alloc((size_t)NODES * sizeof(float));
  float*    sumw  = (float*)alloc((size_t)NODES * sizeof(float));
  float*    lattr = (float*)alloc((size_t)NODES * sizeof(float));
  float*    alpha = (float*)alloc((size_t)EN_TOT * sizeof(float));
  _Float16* wfrag = (_Float16*)alloc((size_t)CIN * CIN * sizeof(_Float16));
  float*    ce    = (float*)alloc(256);

  const int TB = 256;
  auto blk = [](long n) { return (unsigned)((n + 255) / 256); };

  // once per call: f16 staging of x, and self-loop edge-attr mean (layer-independent)
  k_cvt_f16<<<blk((long)NODES * CIN), TB, 0, stream>>>(x, h16, (long)NODES * CIN);
  k_fill_f32<<<blk(NODES), TB, 0, stream>>>(cnt, 0.f, NODES);
  k_fill_f32<<<blk(NODES), TB, 0, stream>>>(sumw, 0.f, NODES);
  k_deg<<<blk(EDGES), TB, 0, stream>>>(dst, ew, cnt, sumw);
  k_loopattr<<<blk(NODES), TB, 0, stream>>>(cnt, sumw, lattr);

  // layer 0: 128->128, edge-attn, relu, result staged to h16
  run_layer<128>(W0, as0, ad0, b0, We0, ae0, 1, 1, src, dst, ew, h16, wfrag, ce,
                 xl, agg, asrc, adstv, menc, den, lattr, alpha, nullptr, h16, stream);
  // layer 1: 128->128, edge-attn, relu, result staged to h16
  run_layer<128>(W1, as1, ad1, b1, We1, ae1, 1, 1, src, dst, ew, h16, wfrag, ce,
                 xl, agg, asrc, adstv, menc, den, lattr, alpha, nullptr, h16, stream);
  // conv_mu: 128->64, edge-attn, no relu, -> d_out[0 : N*64)
  run_layer<64>(Wm, asM, adM, bM, WeM, aeM, 1, 0, src, dst, ew, h16, wfrag, ce,
                xl, agg, asrc, adstv, menc, den, lattr, alpha, outF, nullptr, stream);
  // conv_logstd: 128->64, NO edge term, no relu, -> d_out[N*64 : 2*N*64)
  run_layer<64>(Wl, asL, adL, bL, nullptr, nullptr, 0, 0, src, dst, ew, h16, wfrag, ce,
                xl, agg, asrc, adstv, menc, den, lattr, alpha,
                outF + (size_t)NODES * 64, nullptr, stream);
}
